// ModuleCorrelation_48653389529466
// MI455X (gfx1250) — matrix operand order repaired and checked by reading, not compiled, not run on hardware
//
#include <hip/hip_runtime.h>

typedef __attribute__((ext_vector_type(16))) __bf16 v16bf;
typedef __attribute__((ext_vector_type(8)))  __bf16 v8bf;
typedef __attribute__((ext_vector_type(2)))  __bf16 v2bf;
typedef __attribute__((ext_vector_type(8)))  float  v8f;
typedef __attribute__((ext_vector_type(4)))  float  v4f;

#define C_CH   128
#define HH     112
#define WW     192
#define BB     8
#define DISP   4
#define NDISP  9          // 2*DISP+1 displacement values per axis
#define NCH    81         // NDISP*NDISP output channels
#define LDK    136        // padded K stride in bf16 elems (272B, 16B aligned)

__global__ __launch_bounds__(32)
void corr_wmma_kernel(const float* __restrict__ first,
                      const float* __restrict__ second,
                      float* __restrict__ out)
{
    __shared__ __bf16 A_lds[16 * LDK];   // first tile:  [m (16 px)][c (128)]
    __shared__ __bf16 B_lds[32 * LDK];   // second tile: [n (32 px)][c (128)]

    const int lane = threadIdx.x;        // 0..31 (wave32)
    const int x0 = blockIdx.x * 16;
    const int y  = blockIdx.y;
    const int b  = blockIdx.z;

    const int HW = HH * WW;
    const size_t plane = (size_t)b * C_CH * HW;

    // ---- stage A: first[b, :, y, x0:x0+16] -> A_lds[m][c] as packed bf16 ----
    // per iter: 32 lanes x 8 floats (two channels x 4 x-values) = 256 elems
    {
        const int xl = (lane & 3) * 4;       // 0,4,8,12
        const int c2 = (lane >> 2) * 2;      // 0,2,..,14
        for (int c0 = 0; c0 < C_CH; c0 += 16) {
            const int c = c0 + c2;
            const float* gp0 = first + plane + (size_t)c * HW + (size_t)y * WW + x0 + xl;
            const float* gp1 = gp0 + HW;
            v4f v0 = *(const v4f*)gp0;
            v4f v1 = *(const v4f*)gp1;
#pragma unroll
            for (int j = 0; j < 4; ++j) {
                v2bf p = { (__bf16)v0[j], (__bf16)v1[j] };
                *(v2bf*)&A_lds[(xl + j) * LDK + c] = p;   // ds_store_b32
            }
        }
    }
    __syncthreads();

    // ---- build the 4 A fragments (K = 128 = 4 x 32) ----
    // ISA 16-bit A 16x32 layout: lane L -> M = L%16, khalf = L/16;
    // K runs [khalf*8, +8) and [16+khalf*8, +8) per 32-K step.
    const int m16   = lane & 15;
    const int khalf = lane >> 4;

    v16bf afrag[4];
#pragma unroll
    for (int kk = 0; kk < 4; ++kk) {
        const __bf16* ap = &A_lds[m16 * LDK + kk * 32 + khalf * 8];
        v8bf lo = *(const v8bf*)(ap);        // ds_load_b128
        v8bf hi = *(const v8bf*)(ap + 16);   // ds_load_b128
        afrag[kk] = __builtin_shufflevector(lo, hi,
                        0,1,2,3,4,5,6,7,8,9,10,11,12,13,14,15);
    }

    const float inv_c = 1.0f / (float)C_CH;

    for (int dy = 0; dy < NDISP; ++dy) {
        const int ys = y + dy - DISP;                 // source row in second
        const bool inrow = (ys >= 0) && (ys < HH);    // wave-uniform

        v8f acc[2] = {};                              // two 16-wide N tiles

        if (inrow) {
            // prefetch next dy row of `second` into cache
            if (dy + 1 < NDISP) {
                const int yn = ys + 1;
                if (yn < HH)
                    __builtin_prefetch(second + plane + (size_t)yn * WW + x0, 0, 1);
            }

            // ---- stage B: second[b, :, ys, x0-4 : x0+28] -> B_lds[n][c] ----
            // per iter: 32 lanes x 8 floats (two channels x 4 x-values).
            // x0, DISP, WW are multiples of 4 => each 4-wide x chunk is
            // either fully in-bounds or fully out-of-bounds (never straddles),
            // so OOB handling is just "load or keep zeros".
            {
                const int xl = (lane & 7) * 4;       // 0..28
                const int c2 = (lane >> 3) * 2;      // 0,2,4,6
                const int xg = x0 - DISP + xl;       // multiple of 4
                const bool inx = (xg >= 0) && (xg + 3 < WW);
                for (int c0 = 0; c0 < C_CH; c0 += 8) {
                    const int c = c0 + c2;
                    const float* gp0 = second + plane + (size_t)c * HW
                                       + (size_t)ys * WW + xg;
                    v4f v0 = { 0.0f, 0.0f, 0.0f, 0.0f };
                    v4f v1 = { 0.0f, 0.0f, 0.0f, 0.0f };
                    if (inx) {
                        v0 = *(const v4f*)gp0;
                        v1 = *(const v4f*)(gp0 + HW);
                    }
#pragma unroll
                    for (int j = 0; j < 4; ++j) {
                        v2bf p = { (__bf16)v0[j], (__bf16)v1[j] };
                        *(v2bf*)&B_lds[(xl + j) * LDK + c] = p;  // ds_store_b32
                    }
                }
            }
            __syncthreads();

            // ---- gather B fragments + WMMA, 2 N-tiles x 4 K-steps ----
            // ISA 16-bit B 32x16 layout: lane L -> N = L%16, khalf = L/16;
            // K runs [khalf*8, +8) and [16+khalf*8, +8) per 32-K step.
            v16bf bfrag[2][4];
#pragma unroll
            for (int nt = 0; nt < 2; ++nt) {
                const int n = nt * 16 + m16;
#pragma unroll
                for (int kk = 0; kk < 4; ++kk) {
                    const __bf16* bp = &B_lds[n * LDK + kk * 32 + khalf * 8];
                    v8bf lo = *(const v8bf*)(bp);      // ds_load_b128
                    v8bf hi = *(const v8bf*)(bp + 16); // ds_load_b128
                    bfrag[nt][kk] = __builtin_shufflevector(lo, hi,
                                      0,1,2,3,4,5,6,7,8,9,10,11,12,13,14,15);
                }
            }

#pragma unroll
            for (int kk = 0; kk < 4; ++kk) {
#pragma unroll
                for (int nt = 0; nt < 2; ++nt) {
                    acc[nt] = __builtin_amdgcn_wmma_f32_16x16x32_bf16(
                        /*neg_a=*/false, afrag[kk],
                        /*neg_b=*/false, bfrag[nt][kk],
                        /*c_mod=*/(short)0, acc[nt],
                        /*reuse_a=*/false, /*reuse_b=*/false);
                }
            }
            __syncthreads();   // protect B_lds before next dy overwrites it
        }

        // ---- extract diagonal band (dx in [-4,4]) and store ----
        // D layout: VGPR r, lane L -> M = r + 8*(L/16), N = L%16.
        // x = x0 + M ; dx+4 = nt*16 + N - M.
        // Uniform 64-bit row base + small 32-bit lane offset -> SADDR stores.
        float* const orow = out + (((size_t)b * NCH + (size_t)dy * NDISP) * HH
                                   + y) * WW + x0;
#pragma unroll
        for (int nt = 0; nt < 2; ++nt) {
#pragma unroll
            for (int r = 0; r < 8; ++r) {
                const int M = r + 8 * khalf;
                const int dxp4 = nt * 16 + m16 - M;
                if (dxp4 >= 0 && dxp4 <= 2 * DISP) {
                    orow[dxp4 * HW + M] = acc[nt][r] * inv_c;
                }
            }
        }
    }
}

extern "C" void kernel_launch(void* const* d_in, const int* in_sizes, int n_in,
                              void* d_out, int out_size, void* d_ws, size_t ws_size,
                              hipStream_t stream)
{
    (void)in_sizes; (void)n_in; (void)out_size; (void)d_ws; (void)ws_size;
    const float* first  = (const float*)d_in[0];
    const float* second = (const float*)d_in[1];
    float* out = (float*)d_out;

    dim3 grid(WW / 16, HH, BB);   // (12, 112, 8) = 10752 blocks
    dim3 block(32, 1, 1);         // one wave32 per block
    corr_wmma_kernel<<<grid, block, 0, stream>>>(first, second, out);
}
